// AttnBlock_65231963292122
// MI455X (gfx1250) — matrix operand order repaired and checked by reading, not compiled, not run on hardware
//
#include <hip/hip_runtime.h>
#include <cstdint>
#include <cstddef>

// ---------------------------------------------------------------------------
// MI455X (gfx1250) attention block.
// Precision: bf16 WMMA (v_wmma_f32_16x16x32_bf16) with f32 accumulation.
// All GEMM dims are multiples of 128 (C=512, N=4096) -> no bounds checks.
// GEMM main loops are software-pipelined: double-buffered LDS (ping-pong),
// next tile's global loads issued before the current tile's WMMAs.
// ---------------------------------------------------------------------------

typedef unsigned short ushort_t;
typedef __attribute__((ext_vector_type(16))) __bf16 v16bf;
typedef __attribute__((ext_vector_type(8)))  float  v8f;

union Frag {
    v16bf  v;
    uint4  q[2];
    ushort_t u[16];
};

__device__ __forceinline__ ushort_t f2bf(float f) {
    uint32_t u = __builtin_bit_cast(uint32_t, f);
    u += 0x7FFFu + ((u >> 16) & 1u);   // round-to-nearest-even
    return (ushort_t)(u >> 16);
}

// ---------------------------------------------------------------------------
// GroupNorm: 32 groups, group = 16 channels x 4096 spatial = 65536 contiguous
// floats. grid = B*G = 64 blocks of 256 threads. Output: bf16 h (B,C,N).
// ---------------------------------------------------------------------------
__global__ __launch_bounds__(256) void groupnorm_kernel(
    const float* __restrict__ x, const float* __restrict__ gamma,
    const float* __restrict__ beta, ushort_t* __restrict__ h) {
    constexpr int PER_GROUP = 16 * 4096;
    const int bg  = blockIdx.x;
    const int g   = bg & 31;
    const size_t base = (size_t)bg * PER_GROUP;
    const float* src = x + base;
    const int tid = threadIdx.x;

    float s = 0.f, ss = 0.f;
    for (int i = tid * 4; i < PER_GROUP; i += 1024) {
        float4 v = *(const float4*)(src + i);
        s  += v.x + v.y + v.z + v.w;
        ss += v.x * v.x + v.y * v.y + v.z * v.z + v.w * v.w;
    }
    __shared__ float rs[256], rss[256];
    rs[tid] = s; rss[tid] = ss;
    __syncthreads();
    for (int o = 128; o > 0; o >>= 1) {
        if (tid < o) { rs[tid] += rs[tid + o]; rss[tid] += rss[tid + o]; }
        __syncthreads();
    }
    const float inv  = 1.0f / (float)PER_GROUP;
    const float mean = rs[0] * inv;
    const float var  = rss[0] * inv - mean * mean;
    const float rstd = rsqrtf(var + 1e-5f);

    for (int i = tid * 4; i < PER_GROUP; i += 1024) {
        float4 v = *(const float4*)(src + i);
        const int c = g * 16 + (i >> 12);      // 4096 spatial per channel
        const float ga = gamma[c], be = beta[c];
        union { ushort_t u[4]; uint2 p; } o4;
        o4.u[0] = f2bf((v.x - mean) * rstd * ga + be);
        o4.u[1] = f2bf((v.y - mean) * rstd * ga + be);
        o4.u[2] = f2bf((v.z - mean) * rstd * ga + be);
        o4.u[3] = f2bf((v.w - mean) * rstd * ga + be);
        *(uint2*)(h + base + i) = o4.p;
    }
}

// ---------------------------------------------------------------------------
// Row softmax over 4096 f32, in place. grid = B*N rows, 256 threads.
// ---------------------------------------------------------------------------
__global__ __launch_bounds__(256) void softmax_kernel(float* __restrict__ attn) {
    constexpr int N = 4096;
    float* row = attn + (size_t)blockIdx.x * N;
    const int tid = threadIdx.x;

    float4 v[4];
    float mx = -3.4e38f;
#pragma unroll
    for (int j = 0; j < 4; ++j) {
        v[j] = ((const float4*)row)[tid + j * 256];
        mx = fmaxf(mx, fmaxf(fmaxf(v[j].x, v[j].y), fmaxf(v[j].z, v[j].w)));
    }
    __shared__ float red[256];
    red[tid] = mx; __syncthreads();
    for (int o = 128; o > 0; o >>= 1) {
        if (tid < o) red[tid] = fmaxf(red[tid], red[tid + o]);
        __syncthreads();
    }
    mx = red[0];
    __syncthreads();

    float s = 0.f;
#pragma unroll
    for (int j = 0; j < 4; ++j) {
        v[j].x = __expf(v[j].x - mx); v[j].y = __expf(v[j].y - mx);
        v[j].z = __expf(v[j].z - mx); v[j].w = __expf(v[j].w - mx);
        s += v[j].x + v[j].y + v[j].z + v[j].w;
    }
    red[tid] = s; __syncthreads();
    for (int o = 128; o > 0; o >>= 1) {
        if (tid < o) red[tid] += red[tid + o];
        __syncthreads();
    }
    const float invs = 1.0f / red[0];
#pragma unroll
    for (int j = 0; j < 4; ++j) {
        v[j].x *= invs; v[j].y *= invs; v[j].z *= invs; v[j].w *= invs;
        ((float4*)row)[tid + j * 256] = v[j];
    }
}

// ---------------------------------------------------------------------------
// Tiled bf16 WMMA GEMM: D(M,N) = A(M,K) * B(K,N) [+bias(M)] [*scale] [+res]
//   block = 256 threads (8 waves), tile 128x128, BK=32.
//   wave (wm in 0..3, wn in 0..1) owns 32x64 = 2x4 WMMA tiles.
//   A_F32       : A source is f32 row-major (converted to bf16 in staging)
//   B_TRANS_F32 : B element (k,n) = srcF32[n*ldb + k]  (attn probabilities)
//   OUT_TRANS   : write bf16 D^T (used to produce q^T for the logits GEMM)
//   OUT_BF16    : write bf16 row-major, else f32 row-major
// LDS rows padded to 40 ushorts (80B): 16B-aligned b128 fragment reads,
// conflict-free bank mapping. Double-buffered: 40KB LDS per block.
// Pipeline per iter: [barrier] [glob loads tile k+1 -> regs]
//                    [frag ds_loads + 8x WMMA on tile k] [regs -> LDS buf^1]
// ---------------------------------------------------------------------------
template <bool A_F32, bool B_TRANS_F32, bool OUT_TRANS, bool OUT_BF16, bool USE_SCALE>
__global__ __launch_bounds__(256) void gemm_wmma_kernel(
    const void* __restrict__ Aptr, size_t aBatch, int lda,
    const void* __restrict__ Bptr, size_t bBatch, int ldb,
    void* __restrict__ Optr, size_t oBatch, int ldo,
    const float* __restrict__ bias,
    const float* __restrict__ res, size_t resBatch,
    int K, float scale) {
    constexpr int BM = 128, BN = 128, BK = 32, SRD = 40;
    __shared__ ushort_t As[2][BM][SRD];   // [buf][m][k]
    __shared__ ushort_t Bs[2][BN][SRD];   // [buf][n][k]

    const int tid  = threadIdx.x;
    const int lane = tid & 31, wave = tid >> 5;
    const int wm = wave & 3, wn = wave >> 2;
    const int l15 = lane & 15, lhi = lane >> 4;
    const int m0 = blockIdx.y * BM, n0 = blockIdx.x * BN;
    const int b  = blockIdx.z;

    const float*    Af = (const float*)Aptr + (size_t)b * aBatch;
    const ushort_t* Au = (const ushort_t*)Aptr + (size_t)b * aBatch;
    const float*    Bf = (const float*)Bptr + (size_t)b * bBatch;
    const ushort_t* Bu = (const ushort_t*)Bptr + (size_t)b * bBatch;

    const int ksteps = K / BK;

    // staging registers (next tile)
    float4 afr[2][2]; uint4 aur[2];
    float4 bfr[2][2]; uint4 bur[2];

    auto load_tiles = [&](int ks) {
        const int k0 = ks * BK;
#pragma unroll
        for (int it = 0; it < 2; ++it) {
            const int linear = tid + it * 256;        // 0..511
            const int row = linear >> 2;
            const int kc  = (linear & 3) << 3;
            if constexpr (A_F32) {
                const float* s = Af + (size_t)(m0 + row) * lda + (k0 + kc);
                afr[it][0] = *(const float4*)s;
                afr[it][1] = *(const float4*)(s + 4);
                if (ks + 1 < ksteps) __builtin_prefetch(s + BK, 0, 1);
            } else {
                const ushort_t* s = Au + (size_t)(m0 + row) * lda + (k0 + kc);
                aur[it] = *(const uint4*)s;
                if (ks + 1 < ksteps) __builtin_prefetch(s + BK, 0, 1);
            }
        }
#pragma unroll
        for (int it = 0; it < 2; ++it) {
            const int linear = tid + it * 256;
            if constexpr (B_TRANS_F32) {
                const int n  = linear >> 2;
                const int kg = (linear & 3) << 3;
                const float* s = Bf + (size_t)(n0 + n) * ldb + (k0 + kg);
                bfr[it][0] = *(const float4*)s;
                bfr[it][1] = *(const float4*)(s + 4);
                if (ks + 1 < ksteps) __builtin_prefetch(s + BK, 0, 1);
            } else {
                const int k  = linear >> 4;
                const int ng = (linear & 15) << 3;
                const ushort_t* s = Bu + (size_t)(k0 + k) * ldb + (n0 + ng);
                bur[it] = *(const uint4*)s;
                if (ks + 1 < ksteps) __builtin_prefetch(s + (size_t)BK * ldb, 0, 1);
            }
        }
    };

    auto store_tiles = [&](int buf) {
#pragma unroll
        for (int it = 0; it < 2; ++it) {
            const int linear = tid + it * 256;
            const int row = linear >> 2;
            const int kc  = (linear & 3) << 3;
            if constexpr (A_F32) {
                ushort_t* d = &As[buf][row][kc];
                d[0] = f2bf(afr[it][0].x); d[1] = f2bf(afr[it][0].y);
                d[2] = f2bf(afr[it][0].z); d[3] = f2bf(afr[it][0].w);
                d[4] = f2bf(afr[it][1].x); d[5] = f2bf(afr[it][1].y);
                d[6] = f2bf(afr[it][1].z); d[7] = f2bf(afr[it][1].w);
            } else {
                *(uint4*)&As[buf][row][kc] = aur[it];
            }
        }
#pragma unroll
        for (int it = 0; it < 2; ++it) {
            const int linear = tid + it * 256;
            if constexpr (B_TRANS_F32) {
                const int n  = linear >> 2;
                const int kg = (linear & 3) << 3;
                ushort_t* d = &Bs[buf][n][kg];
                d[0] = f2bf(bfr[it][0].x); d[1] = f2bf(bfr[it][0].y);
                d[2] = f2bf(bfr[it][0].z); d[3] = f2bf(bfr[it][0].w);
                d[4] = f2bf(bfr[it][1].x); d[5] = f2bf(bfr[it][1].y);
                d[6] = f2bf(bfr[it][1].z); d[7] = f2bf(bfr[it][1].w);
            } else {
                const int k  = linear >> 4;
                const int ng = (linear & 15) << 3;
                const ushort_t* u = (const ushort_t*)&bur[it];
#pragma unroll
                for (int j = 0; j < 8; ++j) Bs[buf][ng + j][k] = u[j];
            }
        }
    };

    v8f acc[2][4] = {};

    // prologue: tile 0 -> buf 0
    load_tiles(0);
    store_tiles(0);

    for (int ks = 0; ks < ksteps; ++ks) {
        const int buf = ks & 1;
        __syncthreads();                      // buf ready; buf^1 free

        if (ks + 1 < ksteps) load_tiles(ks + 1);   // global -> regs (overlaps WMMAs)

        // ---- fragments from buf ----
        // A 16x32 bf16 layout: lane<16 holds row=l15, K = {0..7,16..23};
        //                      lane>=16 holds row=l15, K = {8..15,24..31}.
        Frag fa[2], fb[4];
#pragma unroll
        for (int mi = 0; mi < 2; ++mi) {
            const int arow = wm * 32 + mi * 16 + l15;
            const int akb  = lhi * 8;
            fa[mi].q[0] = *(const uint4*)&As[buf][arow][akb];
            fa[mi].q[1] = *(const uint4*)&As[buf][arow][akb + 16];
        }
        // B 32x16 bf16 layout: lane<16 holds col=l15, K = {0..7,16..23}; lane>=16 K = {8..15,24..31}.
#pragma unroll
        for (int ni = 0; ni < 4; ++ni) {
            const int brow = wn * 64 + ni * 16 + l15;
            const int bkb  = lhi * 8;
            fb[ni].q[0] = *(const uint4*)&Bs[buf][brow][bkb];
            fb[ni].q[1] = *(const uint4*)&Bs[buf][brow][bkb + 16];
        }
#pragma unroll
        for (int mi = 0; mi < 2; ++mi)
#pragma unroll
            for (int ni = 0; ni < 4; ++ni)
                acc[mi][ni] = __builtin_amdgcn_wmma_f32_16x16x32_bf16(
                    false, fa[mi].v, false, fb[ni].v,
                    (short)0, acc[mi][ni], false, false);

        if (ks + 1 < ksteps) store_tiles(buf ^ 1);  // regs -> LDS (after WMMAs)
    }

    // ---- epilogue: C/D layout: lane holds col (l15), M = r + 8*lhi ----
#pragma unroll
    for (int mi = 0; mi < 2; ++mi) {
        const int mbase = m0 + wm * 32 + mi * 16 + lhi * 8;
#pragma unroll
        for (int ni = 0; ni < 4; ++ni) {
            const int nG = n0 + wn * 64 + ni * 16 + l15;
            float vals[8];
#pragma unroll
            for (int r = 0; r < 8; ++r) {
                float f = acc[mi][ni][r];
                if constexpr (USE_SCALE) f *= scale;
                if (bias) f += bias[mbase + r];
                vals[r] = f;
            }
            if constexpr (OUT_TRANS) {
                ushort_t* Ot = (ushort_t*)Optr + (size_t)b * oBatch;
                union { ushort_t u[8]; uint4 q; } p;
#pragma unroll
                for (int r = 0; r < 8; ++r) p.u[r] = f2bf(vals[r]);
                *(uint4*)&Ot[(size_t)nG * ldo + mbase] = p.q;  // 16B store
            } else if constexpr (OUT_BF16) {
                ushort_t* O = (ushort_t*)Optr + (size_t)b * oBatch;
#pragma unroll
                for (int r = 0; r < 8; ++r)
                    O[(size_t)(mbase + r) * ldo + nG] = f2bf(vals[r]);
            } else {
                float* O = (float*)Optr + (size_t)b * oBatch;
#pragma unroll
                for (int r = 0; r < 8; ++r) {
                    float f = vals[r];
                    if (res) f += res[(size_t)b * resBatch + (size_t)(mbase + r) * ldo + nG];
                    O[(size_t)(mbase + r) * ldo + nG] = f;
                }
            }
        }
    }
}

// ---------------------------------------------------------------------------
// Orchestration.  Workspace layout:
//   h   bf16 (B,C,N)   8 MiB
//   qt  bf16 (B,N,C)   8 MiB   (q transposed: logits GEMM A operand)
//   kk  bf16 (B,C,N)   8 MiB
//   vv  bf16 (B,C,N)   8 MiB
//   ao  bf16 (B,C,N)   8 MiB   (attn @ v)
//   attn f32 (B,N,N) 128 MiB
// ---------------------------------------------------------------------------
extern "C" void kernel_launch(void* const* d_in, const int* in_sizes, int n_in,
                              void* d_out, int out_size, void* d_ws, size_t ws_size,
                              hipStream_t stream) {
    const float* x     = (const float*)d_in[0];
    const float* gamma = (const float*)d_in[1];
    const float* beta  = (const float*)d_in[2];
    const float* wq = (const float*)d_in[3];  const float* bq = (const float*)d_in[4];
    const float* wk = (const float*)d_in[5];  const float* bk = (const float*)d_in[6];
    const float* wv = (const float*)d_in[7];  const float* bv = (const float*)d_in[8];
    const float* wp = (const float*)d_in[9];  const float* bp = (const float*)d_in[10];

    constexpr int B = 2, C = 512, N = 4096, G = 32;
    const size_t CN = (size_t)C * N;   // 2,097,152
    const size_t NN = (size_t)N * N;   // 16,777,216

    ushort_t* h  = (ushort_t*)d_ws;
    ushort_t* qt = h  + (size_t)B * CN;
    ushort_t* kk = qt + (size_t)B * CN;
    ushort_t* vv = kk + (size_t)B * CN;
    ushort_t* ao = vv + (size_t)B * CN;
    float*  attn = (float*)(ao + (size_t)B * CN);

    const dim3 blk(256);

    // 1) GroupNorm -> h (bf16)
    groupnorm_kernel<<<B * G, blk, 0, stream>>>(x, gamma, beta, h);

    // 2) q/k/v projections (A = f32 weights, B = bf16 h)
    const dim3 gQKV(N / 128, C / 128, B);
    gemm_wmma_kernel<true, false, true,  true,  false><<<gQKV, blk, 0, stream>>>(
        wq, 0, C, h, CN, N, qt, CN, C, bq, nullptr, 0, C, 1.0f);       // q^T (N,C)
    gemm_wmma_kernel<true, false, false, true,  false><<<gQKV, blk, 0, stream>>>(
        wk, 0, C, h, CN, N, kk, CN, N, bk, nullptr, 0, C, 1.0f);       // k (C,N)
    gemm_wmma_kernel<true, false, false, true,  false><<<gQKV, blk, 0, stream>>>(
        wv, 0, C, h, CN, N, vv, CN, N, bv, nullptr, 0, C, 1.0f);       // v (C,N)

    // 3) logits = (q^T k) * C^-0.5  -> f32 (N,N)
    const dim3 gATT(N / 128, N / 128, B);
    gemm_wmma_kernel<false, false, false, false, true><<<gATT, blk, 0, stream>>>(
        qt, CN, C, kk, CN, N, attn, NN, N, nullptr, nullptr, 0, C,
        0.044194173824159216f /* 1/sqrt(512) */);

    // 4) softmax over rows, in place
    softmax_kernel<<<B * N, blk, 0, stream>>>(attn);

    // 5) out(c,i) = sum_j v(c,j) * P(i,j)  (B operand: transposed f32 probs)
    const dim3 gO(N / 128, C / 128, B);
    gemm_wmma_kernel<false, true, false, true, false><<<gO, blk, 0, stream>>>(
        vv, CN, N, attn, NN, N, ao, CN, N, nullptr, nullptr, 0, N, 1.0f);

    // 6) proj + bias + residual -> d_out (f32)
    gemm_wmma_kernel<true, false, false, false, false><<<gO, blk, 0, stream>>>(
        wp, 0, C, ao, CN, N, d_out, CN, N, bp, x, CN, C, 1.0f);
}
